// SVDppModel_24464133718086
// MI455X (gfx1250) — matrix operand order repaired and checked by reading, not compiled, not run on hardware
//
#include <hip/hip_runtime.h>

#define NITEMS 100000
#define BATCH  1024
#define EMBED  64
#define NCHUNK 3125              // NITEMS / 32 (exact)
#define SEG    16                // N-dimension segments
#define WPB    8                 // waves per block (256 threads, wave32)
#define CPW    25                // ceil(NCHUNK / (SEG*WPB))

typedef __attribute__((ext_vector_type(16))) __bf16   v16bf;
typedef __attribute__((ext_vector_type(8)))  float    v8f;
typedef __attribute__((ext_vector_type(4)))  int      i4;
typedef __attribute__((ext_vector_type(4)))  unsigned u4;

union Frag16 { unsigned u[8]; u4 q[2]; v16bf v; };   // 8 VGPRs of packed bf16

__device__ __forceinline__ unsigned f32_to_bf16_rne(float f) {
    unsigned u = __float_as_uint(f);
    return (u + 0x7FFFu + ((u >> 16) & 1u)) >> 16;
}

// pack two mask bits as bf16{0,1} pair: lo half = x, hi half = y
__device__ __forceinline__ unsigned pk2(int x, int y) {
    return (x > 0 ? 0x00003F80u : 0u) | (y > 0 ? 0x3F800000u : 0u);
}

__device__ __forceinline__ void atomAddF(float* p, float v) {
    __hip_atomic_fetch_add(p, v, __ATOMIC_RELAXED, __HIP_MEMORY_SCOPE_AGENT);
}

// ---- kernel 0a: zero acc[1024*64] + cnt[1024] (contiguous) ----------------
__global__ void zero_ws(float* p, int n) {
    int i = blockIdx.x * blockDim.x + threadIdx.x;
    if (i < n) p[i] = 0.0f;
}

// ---- kernel 0b: Y f32 [N,64] -> bf16 B-fragments, per K=32 chunk ----------
// YB flat index i = ((c*4 + t)*32 + lane)*8 + vgpr ; each dword holds
// (K=2v, N=t*16+lane%16) in [15:0] and (K=2v+1, ...) in [31:16], plus +8 for v>=4;
// lanes >=16 carry K offset +8  (32x16 bf16 B layout for wmma 16x16x32).
__global__ void pack_yb(const float* __restrict__ Y, unsigned* __restrict__ YB) {
    int i = blockIdx.x * blockDim.x + threadIdx.x;
    if (i >= NCHUNK * 1024) return;
    int v = i & 7, l = (i >> 3) & 31, t = (i >> 8) & 3, c = i >> 10;
    int k0  = 2 * v + ((l < 16) ? 0 : 8) + ((v < 4) ? 0 : 8);
    int col = t * 16 + (l & 15);
    size_t j = (size_t)c * 32 + k0;
    unsigned lo = f32_to_bf16_rne(Y[j * EMBED + col]);
    unsigned hi = f32_to_bf16_rne(Y[(j + 1) * EMBED + col]);
    YB[i] = lo | (hi << 16);
}

// ---- main: mask @ Y via v_wmma_f32_16x16x32_bf16 --------------------------
__global__ __launch_bounds__(256) void svdpp_mm(
    const int* __restrict__ pos, const unsigned* __restrict__ YB,
    float* __restrict__ acc, float* __restrict__ cnt)
{
    const int lane = threadIdx.x & 31;
    const int wave = threadIdx.x >> 5;
    const int rblk = blockIdx.x;                    // 16-row batch block
    const int widx = blockIdx.y * WPB + wave;       // global wave in this row block
    int c0 = widx * CPW;
    int c1 = c0 + CPW; if (c1 > NCHUNK) c1 = NCHUNK;

    const int    rowm = lane & 15;
    const size_t prow = (size_t)(rblk * 16 + rowm) * NITEMS;
    const int    koff = (lane < 16) ? 0 : 8;        // A-matrix 16x32 bf16 lane layout

    v8f acc0 = {}, acc1 = {}, acc2 = {}, acc3 = {};
    int icnt = 0;

    for (int c = c0; c < c1; ++c) {
        // A: 16 mask values per lane: K = koff..koff+7 and koff+16..koff+23
        const int* p = pos + prow + (size_t)c * 32 + koff;
        i4 a0 = *(const i4*)(p);
        i4 a1 = *(const i4*)(p + 4);
        i4 a2 = *(const i4*)(p + 16);
        i4 a3 = *(const i4*)(p + 20);

        Frag16 A;
        A.u[0] = pk2(a0.x, a0.y); A.u[1] = pk2(a0.z, a0.w);
        A.u[2] = pk2(a1.x, a1.y); A.u[3] = pk2(a1.z, a1.w);
        A.u[4] = pk2(a2.x, a2.y); A.u[5] = pk2(a2.z, a2.w);
        A.u[6] = pk2(a3.x, a3.y); A.u[7] = pk2(a3.z, a3.w);

        icnt += (a0.x > 0) + (a0.y > 0) + (a0.z > 0) + (a0.w > 0)
              + (a1.x > 0) + (a1.y > 0) + (a1.z > 0) + (a1.w > 0)
              + (a2.x > 0) + (a2.y > 0) + (a2.z > 0) + (a2.w > 0)
              + (a3.x > 0) + (a3.y > 0) + (a3.z > 0) + (a3.w > 0);

        // B: pre-packed fragments, 8 dwords per lane per tile; tile stride = 64 uint4
        const u4* yb = (const u4*)(YB + ((size_t)c * 128 + lane) * 8);
        Frag16 B0, B1, B2, B3;
        B0.q[0] = yb[0];   B0.q[1] = yb[1];
        B1.q[0] = yb[64];  B1.q[1] = yb[65];
        B2.q[0] = yb[128]; B2.q[1] = yb[129];
        B3.q[0] = yb[192]; B3.q[1] = yb[193];

        acc0 = __builtin_amdgcn_wmma_f32_16x16x32_bf16(false, A.v, false, B0.v, (short)0, acc0, false, false);
        acc1 = __builtin_amdgcn_wmma_f32_16x16x32_bf16(false, A.v, false, B1.v, (short)0, acc1, false, false);
        acc2 = __builtin_amdgcn_wmma_f32_16x16x32_bf16(false, A.v, false, B2.v, (short)0, acc2, false, false);
        acc3 = __builtin_amdgcn_wmma_f32_16x16x32_bf16(false, A.v, false, B3.v, (short)0, acc3, false, false);
    }

    // positive-count partial for this lane's row (both half-wave lanes cover
    // disjoint K ranges of the same row, so both add)
    atomAddF(&cnt[rblk * 16 + rowm], (float)icnt);

    // C/D 16x16 f32 layout: VGPR r -> M=r (lanes 0-15) / M=r+8 (lanes 16-31), N=lane%16
    const int rlo = (lane < 16) ? 0 : 8;
    #pragma unroll
    for (int r = 0; r < 8; ++r) {
        float* dst = acc + (size_t)(rblk * 16 + rlo + r) * EMBED + (lane & 15);
        atomAddF(dst +  0, acc0[r]);
        atomAddF(dst + 16, acc1[r]);
        atomAddF(dst + 32, acc2[r]);
        atomAddF(dst + 48, acc3[r]);
    }
}

// ---- finalize: out[b] = dot(acc[b]/cnt[b] + U[user], I[item]) + biases ----
__global__ void finalize(const int* __restrict__ user, const int* __restrict__ item,
                         const float* __restrict__ U, const float* __restrict__ I,
                         const float* __restrict__ Ub, const float* __restrict__ Ib,
                         const float* __restrict__ gb,
                         const float* __restrict__ acc, const float* __restrict__ cnt,
                         float* __restrict__ out)
{
    int b = blockIdx.x * blockDim.x + threadIdx.x;
    if (b >= BATCH) return;
    int u = user[b], it = item[b];
    float inv = 1.0f / cnt[b];
    const float* ar = acc + (size_t)b * EMBED;
    const float* ur = U + (size_t)u * EMBED;
    const float* ir = I + (size_t)it * EMBED;
    float s = 0.0f;
    #pragma unroll
    for (int d = 0; d < EMBED; ++d)
        s = fmaf(ar[d] * inv + ur[d], ir[d], s);
    out[b] = s + Ub[u] + Ib[it] + gb[0];
}

extern "C" void kernel_launch(void* const* d_in, const int* in_sizes, int n_in,
                              void* d_out, int out_size, void* d_ws, size_t ws_size,
                              hipStream_t stream) {
    const int*   user = (const int*)d_in[0];
    const int*   item = (const int*)d_in[1];
    const int*   pos  = (const int*)d_in[2];
    const float* U    = (const float*)d_in[3];
    const float* I    = (const float*)d_in[4];
    const float* Y    = (const float*)d_in[5];
    const float* Ub   = (const float*)d_in[6];
    const float* Ib   = (const float*)d_in[7];
    const float* gb   = (const float*)d_in[8];

    // workspace layout: YB (12.8MB) | acc 1024x64 f32 | cnt 1024 f32
    unsigned* YB  = (unsigned*)d_ws;
    float*    acc = (float*)((char*)d_ws + (size_t)NCHUNK * 1024 * 4);
    float*    cnt = acc + BATCH * EMBED;

    hipLaunchKernelGGL(zero_ws, dim3((BATCH * EMBED + BATCH + 255) / 256), dim3(256),
                       0, stream, acc, BATCH * EMBED + BATCH);
    hipLaunchKernelGGL(pack_yb, dim3((NCHUNK * 1024 + 255) / 256), dim3(256),
                       0, stream, Y, YB);
    hipLaunchKernelGGL(svdpp_mm, dim3(64, SEG), dim3(256),
                       0, stream, pos, YB, acc, cnt);
    hipLaunchKernelGGL(finalize, dim3((BATCH + 255) / 256), dim3(256),
                       0, stream, user, item, U, I, Ub, Ib, gb, acc, cnt, (float*)d_out);
}
